// FastTriangle_65335042506787
// MI455X (gfx1250) — matrix operand order repaired and checked by reading, not compiled, not run on hardware
//
#include <hip/hip_runtime.h>
#include <hip/hip_bf16.h>
#include <math.h>

#define N   1024
#define D   64
#define NH  4
#define D4  16
#define DCC 32
#define CC  8
#define EPSV 1e-5f

typedef __attribute__((ext_vector_type(16))) __bf16 v16bf;
typedef __attribute__((ext_vector_type(8)))  __bf16 v8bf;
typedef __attribute__((ext_vector_type(8)))  float  v8f;

union V16BF { v16bf v; v8bf h[2]; };

// ---------------------------------------------------------------------------
// K1: LayerNorm(D=64) + left/right projections. Wave per position.
// lanes 0-15 compute left channels, lanes 16-31 compute right channels.
// left -> p[i][j][0:16], right -> p[j][i][16:32] (transpose fused into store).
// ---------------------------------------------------------------------------
__global__ void k1_ln_proj(const float* __restrict__ pair,
                           const float* __restrict__ ln_g, const float* __restrict__ ln_b,
                           const float* __restrict__ Wl,   const float* __restrict__ Wr,
                           float* __restrict__ p_buf)
{
    const int lane = threadIdx.x & 31;
    const int wave = threadIdx.x >> 5;
    const int col  = lane & 15;
    const float* Wc = (lane < 16) ? Wl : Wr;           // [64][16] row-major
    float w[64];
    #pragma unroll
    for (int k = 0; k < 64; ++k) w[k] = Wc[k * D4 + col];
    const float g0 = ln_g[lane],      b0 = ln_b[lane];
    const float g1 = ln_g[lane + 32], b1 = ln_b[lane + 32];

    const long long base = ((long long)blockIdx.x * 8 + wave) * 4;   // 4 positions/wave
    for (int it = 0; it < 4; ++it) {
        const long long pos = base + it;
        const int i = (int)(pos >> 10), j = (int)(pos & 1023);
        const float x0 = pair[pos * 64 + lane];
        const float x1 = pair[pos * 64 + 32 + lane];
        float s  = x0 + x1;
        float s2 = x0 * x0 + x1 * x1;
        #pragma unroll
        for (int m = 16; m >= 1; m >>= 1) {
            s  += __shfl_xor(s,  m, 32);
            s2 += __shfl_xor(s2, m, 32);
        }
        const float mean = s * (1.0f / 64.0f);
        const float var  = s2 * (1.0f / 64.0f) - mean * mean;
        const float rs   = rsqrtf(var + EPSV);
        const float y0 = (x0 - mean) * rs * g0 + b0;
        const float y1 = (x1 - mean) * rs * g1 + b1;
        float acc = 0.f;
        #pragma unroll
        for (int k = 0; k < 32; ++k) acc += __shfl(y0, k, 32) * w[k];
        #pragma unroll
        for (int k = 0; k < 32; ++k) acc += __shfl(y1, k, 32) * w[32 + k];
        const long long dpos = (lane < 16) ? pos : ((long long)j * N + i);
        p_buf[dpos * DCC + lane] = acc;   // lane<16: ch=lane ; lane>=16: ch=16+col=lane
    }
}

// ---------------------------------------------------------------------------
// K2: attn = tanh(p@Wa+ba)*mask ; value = p@Wv+bv. Stage in LDS, emit bf16
// GEMM operands in WMMA-friendly layouts:
//   attnB[h][i][j]              (A, row-major over j)
//   vT[h][k*CC+c][j]            (B^T, contiguous over j)
// Block tile: 32 rows (a = j of einsum) x 16 cols (b = k of einsum).
// ---------------------------------------------------------------------------
__global__ void k2_attn_value(const float* __restrict__ p_buf,
                              const unsigned char* __restrict__ mask,
                              const float* __restrict__ Wa, const float* __restrict__ ba,
                              const float* __restrict__ Wv, const float* __restrict__ bv,
                              __bf16* __restrict__ attnB, __bf16* __restrict__ vT)
{
    __shared__ __bf16 v_s[DCC][16][32];  // [ch][b_off][a_off]
    __shared__ __bf16 a_s[NH][32][16];   // [h][a_off][b_off]
    const int a0 = blockIdx.y * 32;
    const int b0 = blockIdx.x * 16;
    const int t  = threadIdx.x;

    for (int pp = 0; pp < 2; ++pp) {
        const int posIdx = t + pp * 256;
        const int a_off = posIdx >> 4;
        const int b_off = posIdx & 15;
        const long long pos = (long long)(a0 + a_off) * N + (b0 + b_off);
        float x[DCC];
        #pragma unroll
        for (int k = 0; k < DCC; ++k) x[k] = p_buf[pos * DCC + k];
        const float m = mask[pos] ? 1.0f : 0.0f;
        for (int hh = 0; hh < NH; ++hh) {
            float acc = ba[hh];
            #pragma unroll
            for (int k = 0; k < DCC; ++k) acc += x[k] * Wa[k * NH + hh];
            a_s[hh][a_off][b_off] = (__bf16)(tanhf(acc) * m);
        }
        for (int c = 0; c < DCC; ++c) {
            float acc = bv[c];
            #pragma unroll
            for (int k = 0; k < DCC; ++k) acc += x[k] * Wv[k * DCC + c];
            v_s[c][b_off][a_off] = (__bf16)acc;
        }
    }
    __syncthreads();

    for (int idx = t; idx < DCC * 16 * 32; idx += 256) {
        const int a_off = idx & 31;
        const int b_off = (idx >> 5) & 15;
        const int ch    = idx >> 9;
        const int hh = ch >> 3, c = ch & 7;
        vT[(long long)hh * (N * CC) * N + ((long long)(b0 + b_off) * CC + c) * N + (a0 + a_off)]
            = v_s[ch][b_off][a_off];
    }
    for (int idx = t; idx < NH * 32 * 16; idx += 256) {
        const int b_off = idx & 15;
        const int a_off = (idx >> 4) & 31;
        const int hh    = idx >> 9;
        attnB[(long long)hh * N * N + (long long)(a0 + a_off) * N + (b0 + b_off)]
            = a_s[hh][a_off][b_off];
    }
}

// ---------------------------------------------------------------------------
// K3: per-head GEMM O_h[i, kc] = A_h[i, j] * B_h[j, kc] with bf16 WMMA.
// Block tile 128(i) x 128(kc); 8 waves, each 32(i) x 64(kc) = 2x4 tiles of
// v_wmma_f32_16x16x32_bf16. Fragments loaded directly from global: A lane l
// and l+16 together cover one full 64B line of a row; same for B^T rows.
// L2 (192 MB) holds all heads' operands (72 MB) -> HBM traffic ~compulsory.
// ---------------------------------------------------------------------------
__global__ void k3_einsum_wmma(const __bf16* __restrict__ attnB,
                               const __bf16* __restrict__ vT,
                               float* __restrict__ outp)
{
    const int lane = threadIdx.x & 31;
    const int wave = threadIdx.x >> 5;
    const int hh   = blockIdx.z;
    const int iW   = blockIdx.y * 128 + (wave >> 1) * 32;
    const int kcW  = blockIdx.x * 128 + (wave & 1) * 64;

    const __bf16* A = attnB + (long long)hh * N * N;
    const __bf16* B = vT    + (long long)hh * (N * CC) * N;

    const int r15  = lane & 15;
    const int selA = (lane < 16) ? 0 : 8;    // A: K groups {0..7,16..23} / {8..15,24..31}
    const int selB = (lane < 16) ? 0 : 8;    // B: K groups {0..7,16..23} / {8..15,24..31}

    const __bf16* aRow[2];
    const __bf16* bRow[4];
    #pragma unroll
    for (int ti = 0; ti < 2; ++ti) aRow[ti] = A + (long long)(iW  + ti * 16 + r15) * N;
    #pragma unroll
    for (int tk = 0; tk < 4; ++tk) bRow[tk] = B + (long long)(kcW + tk * 16 + r15) * N;

    v8f acc[2][4];
    #pragma unroll
    for (int ti = 0; ti < 2; ++ti)
        #pragma unroll
        for (int tk = 0; tk < 4; ++tk)
            acc[ti][tk] = (v8f){0.f, 0.f, 0.f, 0.f, 0.f, 0.f, 0.f, 0.f};

    for (int k0 = 0; k0 < N; k0 += 32) {
        V16BF a[2], b[4];
        #pragma unroll
        for (int ti = 0; ti < 2; ++ti) {
            a[ti].h[0] = *(const v8bf*)(aRow[ti] + k0 + selA);
            a[ti].h[1] = *(const v8bf*)(aRow[ti] + k0 + 16 + selA);
        }
        #pragma unroll
        for (int tk = 0; tk < 4; ++tk) {
            b[tk].h[0] = *(const v8bf*)(bRow[tk] + k0 + selB);
            b[tk].h[1] = *(const v8bf*)(bRow[tk] + k0 + 16 + selB);
        }
        #pragma unroll
        for (int ti = 0; ti < 2; ++ti)
            #pragma unroll
            for (int tk = 0; tk < 4; ++tk)
                acc[ti][tk] = __builtin_amdgcn_wmma_f32_16x16x32_bf16(
                    false, a[ti].v, false, b[tk].v, (short)0, acc[ti][tk], false, false);
    }

    // D tile layout: VGPR r holds row (r + 0/8), col = lane&15.
    const int mBase = (lane < 16) ? 0 : 8;
    #pragma unroll
    for (int ti = 0; ti < 2; ++ti) {
        #pragma unroll
        for (int tk = 0; tk < 4; ++tk) {
            const int kc = kcW + tk * 16 + r15;
            const int kk = kc >> 3, cc = kc & 7;
            #pragma unroll
            for (int r = 0; r < 8; ++r) {
                const int row = iW + ti * 16 + mBase + r;
                outp[((long long)row * N + kk) * DCC + hh * CC + cc] = acc[ti][tk][r];
            }
        }
    }
}

// ---------------------------------------------------------------------------
// K4: LayerNorm(32, no affine) + @Wo, in place on d_out. Wave per position.
// ---------------------------------------------------------------------------
__global__ void k4_ln_out(float* __restrict__ io, const float* __restrict__ Wo)
{
    const int lane = threadIdx.x & 31;
    const int wave = threadIdx.x >> 5;
    float w[DCC];
    #pragma unroll
    for (int k = 0; k < DCC; ++k) w[k] = Wo[k * DCC + lane];
    const long long base = ((long long)blockIdx.x * 8 + wave) * 4;
    for (int it = 0; it < 4; ++it) {
        const long long pos = base + it;
        const float x = io[pos * DCC + lane];
        float s = x, s2 = x * x;
        #pragma unroll
        for (int m = 16; m >= 1; m >>= 1) {
            s  += __shfl_xor(s,  m, 32);
            s2 += __shfl_xor(s2, m, 32);
        }
        const float mean = s * (1.0f / 32.0f);
        const float var  = s2 * (1.0f / 32.0f) - mean * mean;
        const float y = (x - mean) * rsqrtf(var + EPSV);
        float acc = 0.f;
        #pragma unroll
        for (int k = 0; k < DCC; ++k) acc += __shfl(y, k, 32) * w[k];
        io[pos * DCC + lane] = acc;
    }
}

extern "C" void kernel_launch(void* const* d_in, const int* in_sizes, int n_in,
                              void* d_out, int out_size, void* d_ws, size_t ws_size,
                              hipStream_t stream)
{
    const float* pair         = (const float*)d_in[0];
    const unsigned char* mask = (const unsigned char*)d_in[1];
    const float* ln_g         = (const float*)d_in[2];
    const float* ln_b         = (const float*)d_in[3];
    const float* Wl           = (const float*)d_in[4];
    const float* Wr           = (const float*)d_in[5];
    const float* Wa           = (const float*)d_in[6];
    const float* ba           = (const float*)d_in[7];
    const float* Wv           = (const float*)d_in[8];
    const float* bv           = (const float*)d_in[9];
    const float* Wo           = (const float*)d_in[10];

    char* ws = (char*)d_ws;
    float*  p_buf = (float*)ws;                                           // 128 MB
    __bf16* attnB = (__bf16*)(ws + (size_t)N * N * DCC * 4);              //   8 MB
    __bf16* vT    = (__bf16*)(ws + (size_t)N * N * DCC * 4
                                 + (size_t)NH * N * N * 2);               //  64 MB
    float* outp = (float*)d_out;

    // 1M positions, 8 waves/block, 4 positions/wave -> 32768 blocks
    k1_ln_proj<<<32768, 256, 0, stream>>>(pair, ln_g, ln_b, Wl, Wr, p_buf);
    // a-tiles 32x, b-tiles 64x
    k2_attn_value<<<dim3(64, 32), 256, 0, stream>>>(p_buf, mask, Wa, ba, Wv, bv, attnB, vT);
    // kc tiles 8192/128=64, i tiles 1024/128=8, heads 4
    k3_einsum_wmma<<<dim3(64, 8, 4), 256, 0, stream>>>(attnB, vT, outp);
    k4_ln_out<<<32768, 256, 0, stream>>>(outp, Wo);
}
